// ActorCritic_87256555585990
// MI455X (gfx1250) — matrix operand order repaired and checked by reading, not compiled, not run on hardware
//
#include <hip/hip_runtime.h>
#include <hip/hip_bf16.h>

#define BB   64
#define AA   512
#define NTK  1000
#define NUS  50
#define DD   512
#define HH   8
#define HDD  64
#define SEQ  513
#define SEQP 544            // padded to multiple of 32 (17 key tiles)

typedef __attribute__((ext_vector_type(8)))  __bf16 v8bf;
typedef __attribute__((ext_vector_type(16))) __bf16 v16bf;
typedef __attribute__((ext_vector_type(8)))  float  v8f;
typedef __attribute__((ext_vector_type(4)))  unsigned int u32x4;
typedef __attribute__((ext_vector_type(8)))  int i32x8;
typedef __attribute__((ext_vector_type(4)))  int i32x4;

__device__ __forceinline__ v16bf ldfrag(const __bf16* p0, const __bf16* p1) {
  v8bf a = *(const v8bf*)p0;
  v8bf b = *(const v8bf*)p1;
  return __builtin_shufflevector(a, b, 0,1,2,3,4,5,6,7,8,9,10,11,12,13,14,15);
}

__device__ __forceinline__ v8f wmma_bf16(v16bf a, v16bf b, v8f c) {
  return __builtin_amdgcn_wmma_f32_16x16x32_bf16(false, a, false, b, (short)0, c, false, false);
}

__device__ __forceinline__ float eluf(float x) {
  return x > 0.f ? x : (__expf(x) - 1.f);
}

// ---------------------------------------------------------------- weights -> bf16
__global__ void conv_bf16_kernel(const float* __restrict__ s, __bf16* __restrict__ d, int n) {
  for (int i = blockIdx.x * blockDim.x + threadIdx.x; i < n; i += gridDim.x * blockDim.x)
    d[i] = (__bf16)s[i];
}

// ---------------------------------------------------------------- build padded seq
__global__ __launch_bounds__(256) void build_seq_kernel(
    const float* __restrict__ usv, const float* __restrict__ task,
    const float* __restrict__ ge, const int* __restrict__ tids,
    const int* __restrict__ uids, float* __restrict__ seqf, __bf16* __restrict__ seqb) {
  int r = blockIdx.x;                // b*SEQP + s
  int b = r / SEQP, s = r % SEQP;
  size_t ob = (size_t)r * DD;
  for (int d = threadIdx.x; d < DD; d += 256) {
    float v = 0.f;
    if (s == 0) {
      v = ge[(size_t)b * DD + d];
    } else if (s <= AA) {
      int a = s - 1;
      if (d < 256) { int t = tids[b * AA + a]; v = task[((size_t)b * NTK + t) * 256 + d]; }
      else         { int u = uids[b * AA + a]; v = usv[((size_t)b * NUS + u) * 256 + (d - 256)]; }
    }
    seqf[ob + d] = v;
    seqb[ob + d] = (__bf16)v;
  }
}

// ---------------------------------------------------------------- generic bf16 WMMA GEMM
// C[M x N] = A[M x K] @ W[K x N] + bias ; tile 64x64, K-step 32, 8 waves.
// AMODE==0: plain row-major A, A-tile staged by the Tensor Data Mover (TDM).
// AMODE==1: A rows assembled as concat(ln[b][1+a][0:512], ln[b][0][0:512]) (actor input)
#define BM 64
#define BN 64
#define BK 32
template <int AMODE, int ACT>
__global__ __launch_bounds__(256) void gemm_bf16_kernel(
    const __bf16* __restrict__ Aw, const __bf16* __restrict__ W,
    const float* __restrict__ bias, __bf16* __restrict__ outb,
    float* __restrict__ outf, int M, int N, int K) {
  __shared__ __bf16 As[BM * BK];   // [64][32] row-major (TDM tile layout)
  __shared__ __bf16 Bs[BN * BK];   // transposed: [n][k]
  int tid  = threadIdx.x;
  int wave = tid >> 5, lane = tid & 31;
  int lh = lane >> 4, lm = lane & 15;
  int m0 = blockIdx.x * BM;
  int n0 = blockIdx.y * BN;
  int wm = (wave & 3) * 16;        // wave M offset
  int wn = (wave >> 2) * 32;       // wave N offset
  v8f c0 = {}; v8f c1 = {};
  int akb0 = lh ? 16 : 0, akb1 = lh ? 24 : 8;
  int bkb  = lh ? 16 : 0;

  // TDM descriptor group1 (invariant across K loop): 2-byte elements,
  // tensor_dim0=32, tensor_dim1=64, tile 32x64, dim0 stride = K elements.
  i32x8 g1;
  g1[0] = (int)(1u << 16);         // data_size = 2B
  g1[1] = (int)((unsigned)BK << 16);   // tensor_dim0[15:0]
  g1[2] = (int)((unsigned)BM << 16);   // tensor_dim0[31:16]=0 | tensor_dim1[15:0]
  g1[3] = (int)((unsigned)BK << 16);   // tensor_dim1[31:16]=0 | tile_dim0
  g1[4] = BM;                      // tile_dim1 | tile_dim2=0
  g1[5] = K;                       // tensor_dim0_stride[31:0]
  g1[6] = 0;
  g1[7] = 0;
  i32x4 gz = {0, 0, 0, 0};
  i32x8 gz8 = {0, 0, 0, 0, 0, 0, 0, 0};
  unsigned lds_as = (unsigned)(size_t)(void*)&As[0];
  unsigned long long abase = (unsigned long long)(size_t)(const void*)Aw;

  for (int kk = 0; kk < K; kk += BK) {
    if (AMODE == 0) {
      // A tile via Tensor Data Mover: wave 0 issues one 2D tile descriptor.
      if (tid < 32) {
        unsigned long long ga = abase + (((unsigned long long)(size_t)m0 * K + kk) << 1);
        u32x4 g0;
        g0[0] = 1u;                               // count=1
        g0[1] = lds_as;                           // lds_addr
        g0[2] = (unsigned)ga;                     // global_addr[31:0]
        g0[3] = (unsigned)(ga >> 32) | (2u << 30);// global_addr[56:32] | type=2
        __builtin_amdgcn_tensor_load_to_lds(g0, g1, gz, gz, gz8, 0);
      }
    } else {
      // actor concat A tile: 256 thr x 16B
      int r  = tid >> 2;
      int cc = (tid & 3) * 8;
      int gr = m0 + r;
      int b = gr >> 9, a = gr & 511;
      int k = kk + cc;
      const __bf16* src = (k < DD) ? Aw + ((size_t)(b * SEQP + 1 + a) * DD + k)
                                   : Aw + ((size_t)(b * SEQP) * DD + (k - DD));
      *(v8bf*)&As[r * BK + cc] = *(const v8bf*)src;
    }
    { // B tile (transposed into LDS)
      int r  = tid >> 3;             // k row 0..31
      int cc = (tid & 7) * 8;        // n 0..56
      v8bf w = *(const v8bf*)(W + (size_t)(kk + r) * N + n0 + cc);
#pragma unroll
      for (int j = 0; j < 8; ++j) Bs[(cc + j) * BK + r] = w[j];
    }
    if (AMODE == 0) {
      __builtin_amdgcn_s_wait_tensorcnt(0);
      if (kk + BK < K)               // gfx1250 global_prefetch path for W
        __builtin_prefetch(W + (size_t)(kk + BK + (tid >> 3)) * N + n0, 0, 0);
    }
    __syncthreads();

    v16bf af  = ldfrag(&As[(wm + lm) * BK + akb0], &As[(wm + lm) * BK + akb1]);
    v16bf bf0 = ldfrag(&Bs[(wn + lm) * BK + bkb], &Bs[(wn + lm) * BK + bkb + 8]);
    v16bf bf1 = ldfrag(&Bs[(wn + 16 + lm) * BK + bkb], &Bs[(wn + 16 + lm) * BK + bkb + 8]);
    c0 = wmma_bf16(af, bf0, c0);
    c1 = wmma_bf16(af, bf1, c1);
    __syncthreads();
  }

  float b0v = bias ? bias[n0 + wn + lm] : 0.f;
  float b1v = bias ? bias[n0 + wn + 16 + lm] : 0.f;
#pragma unroll
  for (int i = 0; i < 8; ++i) {
    int row = m0 + wm + lh * 8 + i;
    float x0 = c0[i] + b0v;
    float x1 = c1[i] + b1v;
    if (ACT) { x0 = eluf(x0); x1 = eluf(x1); }
    size_t o0 = (size_t)row * N + n0 + wn + lm;
    size_t o1 = o0 + 16;
    if (outb) { outb[o0] = (__bf16)x0; outb[o1] = (__bf16)x1; }
    if (outf) { outf[o0] = x0;         outf[o1] = x1; }
  }
}

// ---------------------------------------------------------------- V transpose (b,s,h,hd) -> (b,h,hd,s)
__global__ void transpose_v_kernel(const __bf16* __restrict__ v, __bf16* __restrict__ vT, int total) {
  for (int i = blockIdx.x * blockDim.x + threadIdx.x; i < total; i += gridDim.x * blockDim.x) {
    int hd = i & 63;
    int h  = (i >> 6) & 7;
    int rem = i >> 9;
    int s = rem % SEQP;
    int b = rem / SEQP;
    vT[(((size_t)b * HH + h) * HDD + hd) * SEQP + s] = v[i];
  }
}

// ---------------------------------------------------------------- flash attention, 1 wave per (b,h,16-q tile)
__global__ __launch_bounds__(128) void attn_kernel(
    const __bf16* __restrict__ q, const __bf16* __restrict__ k,
    const __bf16* __restrict__ vT, __bf16* __restrict__ ctx) {
  __shared__ __bf16 Ps[4][16 * 32];
  int tid = threadIdx.x;
  int wave = tid >> 5, lane = tid & 31;
  int lh = lane >> 4, lm = lane & 15;
  int wv = blockIdx.x * 4 + wave;
  const int QT = SEQP / 16;                 // 34
  int qt = wv % QT;
  int h  = (wv / QT) & 7;
  int b  = wv / (QT * HH);
  int q0 = qt * 16;

  int akb0 = lh ? 16 : 0, akb1 = lh ? 24 : 8;
  int bkb  = lh ? 16 : 0;

  size_t qbase = ((size_t)b * SEQP + q0 + lm) * DD + h * HDD;
  v16bf aq0 = ldfrag(q + qbase + akb0,      q + qbase + akb1);       // hd 0..31
  v16bf aq1 = ldfrag(q + qbase + 32 + akb0, q + qbase + 32 + akb1);  // hd 32..63

  v8f o0 = {}, o1 = {}, o2 = {}, o3 = {};
  float mrow[8], lrow[8];
#pragma unroll
  for (int i = 0; i < 8; ++i) { mrow[i] = -3e38f; lrow[i] = 0.f; }

  for (int kt = 0; kt < SEQP / 32; ++kt) {  // 17 key tiles of 32
    int kb = kt * 32;
    size_t kbl = ((size_t)b * SEQP + kb + lm) * DD + h * HDD;
    size_t kbh = ((size_t)b * SEQP + kb + 16 + lm) * DD + h * HDD;
    v16bf kf00 = ldfrag(k + kbl + bkb,      k + kbl + bkb + 8);
    v16bf kf01 = ldfrag(k + kbl + 32 + bkb, k + kbl + 32 + bkb + 8);
    v16bf kf10 = ldfrag(k + kbh + bkb,      k + kbh + bkb + 8);
    v16bf kf11 = ldfrag(k + kbh + 32 + bkb, k + kbh + 32 + bkb + 8);
    v8f s0 = {}, s1 = {};
    s0 = wmma_bf16(aq0, kf00, s0); s0 = wmma_bf16(aq1, kf01, s0);
    s1 = wmma_bf16(aq0, kf10, s1); s1 = wmma_bf16(aq1, kf11, s1);

    float mk0 = ((kb + lm)      < SEQ) ? 0.f : -1e30f;
    float mk1 = ((kb + 16 + lm) < SEQ) ? 0.f : -1e30f;
    float rmax[8];
#pragma unroll
    for (int i = 0; i < 8; ++i) {
      float a = s0[i] * 0.125f + mk0; s0[i] = a;
      float c = s1[i] * 0.125f + mk1; s1[i] = c;
      float t = fmaxf(a, c);
      for (int j = 1; j < 16; j <<= 1) t = fmaxf(t, __shfl_xor(t, j, 16));
      rmax[i] = t;
    }
#pragma unroll
    for (int i = 0; i < 8; ++i) {
      float mn = fmaxf(mrow[i], rmax[i]);
      float corr = __expf(mrow[i] - mn);
      mrow[i] = mn;
      float p0 = __expf(s0[i] - mn);
      float p1 = __expf(s1[i] - mn);
      float ps = p0 + p1;
      for (int j = 1; j < 16; j <<= 1) ps += __shfl_xor(ps, j, 16);
      lrow[i] = lrow[i] * corr + ps;
      o0[i] *= corr; o1[i] *= corr; o2[i] *= corr; o3[i] *= corr;
      int m = i + lh * 8;
      Ps[wave][m * 32 + lm]      = (__bf16)p0;
      Ps[wave][m * 32 + 16 + lm] = (__bf16)p1;
    }
    v16bf ap = ldfrag(&Ps[wave][lm * 32 + akb0], &Ps[wave][lm * 32 + akb1]);
    size_t vb0 = (((size_t)b * HH + h) * HDD + 0  + lm) * SEQP + kb + bkb;
    size_t vb1 = (((size_t)b * HH + h) * HDD + 16 + lm) * SEQP + kb + bkb;
    size_t vb2 = (((size_t)b * HH + h) * HDD + 32 + lm) * SEQP + kb + bkb;
    size_t vb3 = (((size_t)b * HH + h) * HDD + 48 + lm) * SEQP + kb + bkb;
    o0 = wmma_bf16(ap, ldfrag(vT + vb0, vT + vb0 + 8), o0);
    o1 = wmma_bf16(ap, ldfrag(vT + vb1, vT + vb1 + 8), o1);
    o2 = wmma_bf16(ap, ldfrag(vT + vb2, vT + vb2 + 8), o2);
    o3 = wmma_bf16(ap, ldfrag(vT + vb3, vT + vb3 + 8), o3);
  }
#pragma unroll
  for (int i = 0; i < 8; ++i) {
    float inv = lrow[i] > 0.f ? 1.f / lrow[i] : 0.f;
    int s = q0 + i + lh * 8;
    size_t ob = ((size_t)b * SEQP + s) * DD + h * HDD;
    ctx[ob + lm]      = (__bf16)(o0[i] * inv);
    ctx[ob + 16 + lm] = (__bf16)(o1[i] * inv);
    ctx[ob + 32 + lm] = (__bf16)(o2[i] * inv);
    ctx[ob + 48 + lm] = (__bf16)(o3[i] * inv);
  }
}

// ---------------------------------------------------------------- residual + layernorm -> bf16
__global__ __launch_bounds__(256) void ln_kernel(
    const float* __restrict__ seqf, const float* __restrict__ ao,
    const float* __restrict__ g, const float* __restrict__ bta, __bf16* __restrict__ lnb) {
  int r = blockIdx.x;                      // 0 .. B*513-1
  int b = r / SEQ, s = r % SEQ;
  size_t base = ((size_t)b * SEQP + s) * DD;
  int tid = threadIdx.x;
  float x0 = seqf[base + tid]       + ao[base + tid];
  float x1 = seqf[base + tid + 256] + ao[base + tid + 256];
  float sum = x0 + x1, sq = x0 * x0 + x1 * x1;
  for (int o = 16; o > 0; o >>= 1) { sum += __shfl_xor(sum, o, 32); sq += __shfl_xor(sq, o, 32); }
  __shared__ float ws[17];
  int wave = tid >> 5, lane = tid & 31;
  if (lane == 0) { ws[wave] = sum; ws[8 + wave] = sq; }
  __syncthreads();
  if (wave == 0 && lane < 8) {
    sum = ws[lane]; sq = ws[8 + lane];
    for (int o = 4; o > 0; o >>= 1) { sum += __shfl_xor(sum, o, 8); sq += __shfl_xor(sq, o, 8); }
    if (lane == 0) { ws[0] = sum; ws[1] = sq; }
  }
  __syncthreads();
  float mean = ws[0] * (1.f / DD);
  float var  = ws[1] * (1.f / DD) - mean * mean;
  float rst  = rsqrtf(var + 1e-5f);
  lnb[base + tid]       = (__bf16)((x0 - mean) * rst * g[tid]       + bta[tid]);
  lnb[base + tid + 256] = (__bf16)((x1 - mean) * rst * g[tid + 256] + bta[tid + 256]);
}

// ---------------------------------------------------------------- actor tail: 64 -> ELU(32) -> 1
__global__ __launch_bounds__(256) void actor_tail_kernel(
    const __bf16* __restrict__ h1, const float* __restrict__ aw2, const float* __restrict__ ab2,
    const float* __restrict__ aw3, const float* __restrict__ ab3, float* __restrict__ out) {
  int r = blockIdx.x * blockDim.x + threadIdx.x;      // 0..32767
  if (r >= BB * AA) return;
  float hin[64];
#pragma unroll
  for (int kk = 0; kk < 64; ++kk) hin[kk] = (float)h1[(size_t)r * 64 + kk];
  float acc = ab3[0];
  for (int j = 0; j < 32; ++j) {
    float a = ab2[j];
    for (int kk = 0; kk < 64; ++kk) a += hin[kk] * aw2[kk * 32 + j];
    acc += eluf(a) * aw3[j];
  }
  out[r] = acc;
}

// ---------------------------------------------------------------- critic MLP
__global__ void critic_kernel(
    const float* __restrict__ ge, const float* __restrict__ cw1, const float* __restrict__ cb1,
    const float* __restrict__ cw2, const float* __restrict__ cb2,
    const float* __restrict__ cw3, const float* __restrict__ cb3, float* __restrict__ out) {
  int b = blockIdx.x * blockDim.x + threadIdx.x;
  if (b >= BB) return;
  float h1v[64];
  for (int j = 0; j < 64; ++j) {
    float a = cb1[j];
    for (int kk = 0; kk < DD; ++kk) a += ge[(size_t)b * DD + kk] * cw1[kk * 64 + j];
    h1v[j] = eluf(a);
  }
  float acc = cb3[0];
  for (int j = 0; j < 32; ++j) {
    float a = cb2[j];
    for (int kk = 0; kk < 64; ++kk) a += h1v[kk] * cw2[kk * 32 + j];
    acc += eluf(a) * cw3[j];
  }
  out[BB * AA + b] = acc;
}

// ================================================================ host
extern "C" void kernel_launch(void* const* d_in, const int* in_sizes, int n_in,
                              void* d_out, int out_size, void* d_ws, size_t ws_size,
                              hipStream_t stream) {
  (void)in_sizes; (void)n_in; (void)out_size; (void)ws_size;
  const float* usv  = (const float*)d_in[0];
  const float* task = (const float*)d_in[1];
  const float* ge   = (const float*)d_in[2];
  const float* wq   = (const float*)d_in[3];
  const float* bq   = (const float*)d_in[4];
  const float* wk   = (const float*)d_in[5];
  const float* bk   = (const float*)d_in[6];
  const float* wvv  = (const float*)d_in[7];
  const float* bv   = (const float*)d_in[8];
  const float* wo   = (const float*)d_in[9];
  const float* bo   = (const float*)d_in[10];
  const float* lng  = (const float*)d_in[11];
  const float* lnbt = (const float*)d_in[12];
  const float* aw1  = (const float*)d_in[13];
  const float* ab1  = (const float*)d_in[14];
  const float* aw2  = (const float*)d_in[15];
  const float* ab2  = (const float*)d_in[16];
  const float* aw3  = (const float*)d_in[17];
  const float* ab3  = (const float*)d_in[18];
  const float* cw1  = (const float*)d_in[19];
  const float* cb1  = (const float*)d_in[20];
  const float* cw2  = (const float*)d_in[21];
  const float* cb2  = (const float*)d_in[22];
  const float* cw3  = (const float*)d_in[23];
  const float* cb3  = (const float*)d_in[24];
  const int* tids   = (const int*)d_in[25];
  const int* uids   = (const int*)d_in[26];
  float* out = (float*)d_out;

  const size_t N1 = (size_t)BB * SEQP * DD;     // 17,825,792
  char* p = (char*)d_ws;
  float*  seqf = (float*)p;        p += N1 * 4;
  __bf16* seqb = (__bf16*)p;       p += N1 * 2;
  __bf16* qb   = (__bf16*)p;       p += N1 * 2;
  __bf16* kb   = (__bf16*)p;       p += N1 * 2;
  __bf16* vb   = (__bf16*)p;       p += N1 * 2;
  __bf16* vT   = (__bf16*)p;       p += N1 * 2;
  __bf16* ctx  = (__bf16*)p;       p += N1 * 2;
  float*  ao   = (float*)p;        p += N1 * 4;
  __bf16* lnb  = (__bf16*)p;       p += N1 * 2;
  __bf16* h1   = (__bf16*)p;       p += (size_t)BB * AA * 64 * 2;
  __bf16* wqb  = (__bf16*)p;       p += (size_t)DD * DD * 2;
  __bf16* wkb  = (__bf16*)p;       p += (size_t)DD * DD * 2;
  __bf16* wvb  = (__bf16*)p;       p += (size_t)DD * DD * 2;
  __bf16* wob  = (__bf16*)p;       p += (size_t)DD * DD * 2;
  __bf16* aw1b = (__bf16*)p;       p += (size_t)2 * DD * 64 * 2;

  // weights -> bf16
  conv_bf16_kernel<<<512, 256, 0, stream>>>(wq,  wqb,  DD * DD);
  conv_bf16_kernel<<<512, 256, 0, stream>>>(wk,  wkb,  DD * DD);
  conv_bf16_kernel<<<512, 256, 0, stream>>>(wvv, wvb,  DD * DD);
  conv_bf16_kernel<<<512, 256, 0, stream>>>(wo,  wob,  DD * DD);
  conv_bf16_kernel<<<256, 256, 0, stream>>>(aw1, aw1b, 2 * DD * 64);

  // gather + pad
  build_seq_kernel<<<BB * SEQP, 256, 0, stream>>>(usv, task, ge, tids, uids, seqf, seqb);

  // QKV projections
  const int M = BB * SEQP;                       // 34816
  dim3 gqkv(M / BM, DD / BN);
  gemm_bf16_kernel<0,0><<<gqkv, 256, 0, stream>>>(seqb, wqb, bq, qb, nullptr, M, DD, DD);
  gemm_bf16_kernel<0,0><<<gqkv, 256, 0, stream>>>(seqb, wkb, bk, kb, nullptr, M, DD, DD);
  gemm_bf16_kernel<0,0><<<gqkv, 256, 0, stream>>>(seqb, wvb, bv, vb, nullptr, M, DD, DD);

  // V transpose for P@V fragments
  transpose_v_kernel<<<8192, 256, 0, stream>>>(vb, vT, (int)N1);

  // attention: one wave per (b, h, 16-query tile)
  attn_kernel<<<(BB * HH * (SEQP / 16)) / 4, 128, 0, stream>>>(qb, kb, vT, ctx);

  // output projection (f32 out for residual)
  gemm_bf16_kernel<0,0><<<gqkv, 256, 0, stream>>>(ctx, wob, bo, nullptr, ao, M, DD, DD);

  // residual + layernorm -> bf16
  ln_kernel<<<BB * SEQ, 256, 0, stream>>>(seqf, ao, lng, lnbt, lnb);

  // actor layer 1: concat(action, state) @ aw1, ELU, K=1024, N=64
  dim3 ga1((BB * AA) / BM, 1);
  gemm_bf16_kernel<1,1><<<ga1, 256, 0, stream>>>(lnb, aw1b, ab1, h1, nullptr, BB * AA, 64, 2 * DD);

  // actor tail + critic
  actor_tail_kernel<<<(BB * AA) / 256, 256, 0, stream>>>(h1, aw2, ab2, aw3, ab3, out);
  critic_kernel<<<1, 64, 0, stream>>>(ge, cw1, cb1, cw2, cb2, cw3, cb3, out);
}
